// Generator_6176162972222
// MI455X (gfx1250) — matrix operand order repaired and checked by reading, not compiled, not run on hardware
//
#include <hip/hip_runtime.h>
#include <hip/hip_bf16.h>

// CDNA5 / gfx1250, wave32. All matrix math through v_wmma_f32_16x16x32_f16
// (f16 operands, fp32 accumulate). Attention fused flash-style to avoid
// materializing the 8x4096x4096 attention matrix (512MB of HBM traffic).
// LDS tiles laid out so every WMMA fragment is two contiguous 16B runs
// (ds_load_b128); staging vectorized; im2col gather uses clamped-offset +
// value-select; softmax reciprocal hoisted out of the streaming loop.

typedef __attribute__((ext_vector_type(16))) _Float16 v16h;
typedef __attribute__((ext_vector_type(8)))  _Float16 v8h;
typedef __attribute__((ext_vector_type(4)))  _Float16 v4h;
typedef __attribute__((ext_vector_type(8)))  float    v8f;

#define WMMA(a, b, c) __builtin_amdgcn_wmma_f32_16x16x32_f16( \
    false, (a), false, (b), (short)0, (c), false, false)

static __device__ __forceinline__ v8h ld8(const _Float16* p) {
  return *(const v8h*)p;                       // 16B-aligned by construction
}
static __device__ __forceinline__ v16h cat8(v8h lo, v8h hi) {
  return __builtin_shufflevector(lo, hi, 0, 1, 2, 3, 4, 5, 6, 7,
                                 8, 9, 10, 11, 12, 13, 14, 15);
}
static __device__ __forceinline__ v4h cvt4(float4 f) {
  v4h h;
  h[0] = (_Float16)f.x; h[1] = (_Float16)f.y;
  h[2] = (_Float16)f.z; h[3] = (_Float16)f.w;
  return h;
}

// ---------------------------------------------------------------- broadcast
__global__ void k_bcast(const float* __restrict__ src, float* __restrict__ dst,
                        int n_per_b, int total) {
  int i = blockIdx.x * blockDim.x + threadIdx.x;
  if (i < total) dst[i] = src[i % n_per_b];
}

// ---------------------------------------------------------------- WMMA GEMM
// Y[b, o, hw] = sum_k A[o, k] * Bmat(k, col=b*HW+hw) + bias[o]
// Block: 128 threads = 4 waves; tile M=16, N=64. Requires M%16==0,
// (B*HW)%64==0, K%32==0 (true for every call in this net).
template <bool IM2COL>
__global__ __launch_bounds__(128) void k_gemm(
    const float* __restrict__ A, const float* __restrict__ X,
    const float* __restrict__ bias, float* __restrict__ Y,
    int M, int K, int HW, int H, int W, int Cin) {
  __shared__ alignas(16) _Float16 At[16][32];     // [m][k] : A-frag rows contiguous
  __shared__ alignas(16) _Float16 Bt[4][16][32];  // [wave][n][k] : transposed
  const int tid  = threadIdx.x;
  const int wv   = tid >> 5;
  const int lane = tid & 31;
  const int half = lane >> 4;
  const int l16  = lane & 15;
  const int m0   = blockIdx.x * 16;
  const int n0   = blockIdx.y * 64;
  const int am    = tid >> 3;            // A stage: row
  const int ak4   = (tid & 7) * 4;       // A stage: 4-wide k offset
  const int nloc  = tid & 15;            // B stage: fixed n
  const int kbase = (tid >> 4) & 7;      // B stage: kk = kbase + ukk*8

  // hoisted per-section (u>>2) column coordinates — no divisions in K loop
  int hcoord[4], wcoord[4];
  const float* xbase[4];
#pragma unroll
  for (int uw = 0; uw < 4; ++uw) {
    int col = n0 + uw * 16 + nloc;
    int b = col / HW, hw = col % HW;
    if (IM2COL) {
      hcoord[uw] = hw / W;
      wcoord[uw] = hw % W;
      xbase[uw]  = X + b * (Cin * H * W);          // 32-bit offsets below
    } else {
      xbase[uw]  = X + b * (K * HW) + hw;
    }
  }
  // incremental k -> (ic = k/9, r = k%9): one division per slot per kernel
  int icv[4], rv[4];
#pragma unroll
  for (int ukk = 0; ukk < 4; ++ukk) {
    int kf = kbase + ukk * 8;
    icv[ukk] = kf / 9;
    rv[ukk]  = kf % 9;
  }
  const float* aptr = A + (m0 + am) * K + ak4;

  v8f acc = {};
  for (int k0 = 0; k0 < K; k0 += 32) {
    // stage A: one float4 load + one b64 LDS store per thread
    *(v4h*)&At[am][ak4] = cvt4(*(const float4*)aptr);
    aptr += 32;
    // stage B: 16 elements per thread, hoisted coordinates
#pragma unroll
    for (int ukk = 0; ukk < 4; ++ukk) {
      int kk = kbase + ukk * 8;
      if (IM2COL) {
        int dy = rv[ukk] / 3 - 1, dx = rv[ukk] % 3 - 1;
        int crow = icv[ukk] * H;                   // 32-bit
#pragma unroll
        for (int uw = 0; uw < 4; ++uw) {
          int hh = hcoord[uw] + dy, ww = wcoord[uw] + dx;
          bool inb = (hh >= 0) & (hh < H) & (ww >= 0) & (ww < W);
          int off  = inb ? ((crow + hh) * W + ww) : 0;   // clamp: off 0 valid
          float t  = xbase[uw][off];                     // unconditional load
          Bt[uw][nloc][kk] = (_Float16)(inb ? t : 0.0f);
        }
      } else {
        int off = (k0 + kk) * HW;                  // 32-bit
#pragma unroll
        for (int uw = 0; uw < 4; ++uw) {
          Bt[uw][nloc][kk] = (_Float16)xbase[uw][off];
        }
      }
    }
    if (IM2COL) {
#pragma unroll
      for (int ukk = 0; ukk < 4; ++ukk) {          // advance (ic, r) by k += 32
        rv[ukk] += 5;
        icv[ukk] += 3;
        if (rv[ukk] >= 9) { rv[ukk] -= 9; icv[ukk] += 1; }
      }
    }
    __syncthreads();
    v16h a  = cat8(ld8(&At[l16][half * 8]), ld8(&At[l16][16 + half * 8]));
    v16h bf = cat8(ld8(&Bt[wv][l16][half * 8]), ld8(&Bt[wv][l16][16 + half * 8]));
    acc = WMMA(a, bf, acc);
    __syncthreads();
  }
  int col = n0 + wv * 16 + l16;
  int b = col / HW, hw = col % HW;
#pragma unroll
  for (int v = 0; v < 8; ++v) {
    int o = m0 + v + 8 * half;                     // C/D row = v + 8*half
    Y[(b * M + o) * HW + hw] = acc[v] + bias[o];
  }
}

// ---------------------------------------------------------------- attention
// out[b, c, i] = sum_j v[b,c,j] * softmax_j( sum_d q[b,d,i] k[b,d,j] )
// One workgroup = (b, 16-query tile). 8 waves; wave w owns channels
// [w*C/8, (w+1)*C/8). Phase 1: online softmax stats. Phase 2: stream 32-key
// chunks; s = q^T k via WMMA; P = exp(s-m)*(1/l) transposed through per-wave
// LDS into a B-fragment; out += V_tile x P^T via WMMA.
__global__ __launch_bounds__(256) void k_attn(
    const float* __restrict__ q, const float* __restrict__ k,
    const float* __restrict__ v, float* __restrict__ out,
    int C, int d, int HW) {
  __shared__ alignas(16) _Float16 qf[16][64];      // [i][c], zero-padded c->64
  __shared__ alignas(16) _Float16 kct[32][64];     // [j][c], zero-padded
  __shared__ alignas(16) _Float16 vc[512][32];     // [c][j]
  __shared__ alignas(16) _Float16 tbt[8][16][32];  // per-wave P staging [i][j]
  __shared__ float red[8][32];
  __shared__ float mrow[16], lrow[16];             // lrow holds 1/l

  const int tid  = threadIdx.x;
  const int wv   = tid >> 5;
  const int lane = tid & 31;
  const int half = lane >> 4;
  const int l16  = lane & 15;
  const int b    = blockIdx.y;
  const int i0   = blockIdx.x * 16;

#pragma unroll
  for (int u = 0; u < 4; ++u) {                             // stage q (coalesced in i)
    int idx = tid + u * 256;
    int ii = idx & 15, c = idx >> 4;
    qf[ii][c] = (c < d) ? (_Float16)q[(b * d + c) * HW + i0 + ii] : (_Float16)0.0f;
  }
  __syncthreads();

  // ---- phase 1: per-query max & sum-exp (online, then shuffle+LDS reduce)
  float mloc[16], lloc[16];
#pragma unroll
  for (int i = 0; i < 16; ++i) { mloc[i] = -1e30f; lloc[i] = 0.0f; }
  for (int j = tid; j < HW; j += 256) {
    float s[16];
#pragma unroll
    for (int i = 0; i < 16; ++i) s[i] = 0.0f;
    for (int c = 0; c < d; ++c) {
      float kv = k[(b * d + c) * HW + j];
#pragma unroll
      for (int i = 0; i < 16; ++i) s[i] += (float)qf[i][c] * kv;
    }
#pragma unroll
    for (int i = 0; i < 16; ++i) {
      float nm = fmaxf(mloc[i], s[i]);
      lloc[i] = lloc[i] * __expf(mloc[i] - nm) + __expf(s[i] - nm);
      mloc[i] = nm;
    }
  }
#pragma unroll
  for (int off = 16; off > 0; off >>= 1) {
#pragma unroll
    for (int i = 0; i < 16; ++i) {
      float om = __shfl_xor(mloc[i], off, 32);
      float ol = __shfl_xor(lloc[i], off, 32);
      float nm = fmaxf(mloc[i], om);
      lloc[i] = lloc[i] * __expf(mloc[i] - nm) + ol * __expf(om - nm);
      mloc[i] = nm;
    }
  }
  if (lane < 16) { red[wv][lane] = mloc[lane]; red[wv][16 + lane] = lloc[lane]; }
  __syncthreads();
  if (tid < 16) {
    float nm = -1e30f;
#pragma unroll
    for (int w = 0; w < 8; ++w) nm = fmaxf(nm, red[w][tid]);
    float l = 0.0f;
#pragma unroll
    for (int w = 0; w < 8; ++w) l += red[w][16 + tid] * __expf(red[w][tid] - nm);
    mrow[tid] = nm;
    lrow[tid] = 1.0f / l;                // reciprocal ONCE, not per chunk
  }
  __syncthreads();

  // ---- A fragments for s = q^T k (invariant across chunks)
  v16h a0 = cat8(ld8(&qf[l16][half * 8]), ld8(&qf[l16][16 + half * 8]));
  v16h a1 = cat8(ld8(&qf[l16][32 + half * 8]), ld8(&qf[l16][48 + half * 8]));

  const int ntile = C >> 7;            // c-tiles per wave: C/128 in {1,2,4}
  const int cbase = wv * (C >> 3);
  const int vit   = C >> 5;            // v-staging iterations (4-wide)
  v8f zero = {};
  v8f acc[4];
#pragma unroll
  for (int t = 0; t < 4; ++t) acc[t] = zero;

  // ---- phase 2: stream key/value chunks of 32
  for (int j0 = 0; j0 < HW; j0 += 32) {
    int jn = HW - j0; if (jn > 32) jn = 32;
#pragma unroll
    for (int u = 0; u < 8; ++u) {                           // stage k (transposed)
      int idx = tid + u * 256;
      int jj = idx & 31, c = idx >> 5;
      kct[jj][c] = (c < d && jj < jn)
                       ? (_Float16)k[(b * d + c) * HW + j0 + jj] : (_Float16)0.0f;
    }
    for (int u = 0; u < vit; ++u) {                         // stage v: float4 -> b64
      int ch = tid + u * 256;                               // 4-elem chunk id
      int c = ch >> 3, jj4 = (ch & 7) * 4;
      v4h h = {};
      if (jj4 < jn)
        h = cvt4(*(const float4*)&v[(b * C + c) * HW + j0 + jj4]);
      *(v4h*)&vc[c][jj4] = h;
    }
    __syncthreads();

#pragma unroll
    for (int hj = 0; hj < 2; ++hj) {                        // two 16-key halves
      const _Float16* kr = &kct[hj * 16 + l16][0];
      v16h b0 = cat8(ld8(kr + half * 8), ld8(kr + 16 + half * 8));
      v16h b1 = cat8(ld8(kr + 32 + half * 8), ld8(kr + 48 + half * 8));
      v8f s = zero;
      s = WMMA(a0, b0, s);
      if (d > 32) s = WMMA(a1, b1, s);
      int jcol = j0 + hj * 16 + l16;
#pragma unroll
      for (int vv = 0; vv < 8; ++vv) {                      // P = exp(s-m)*(1/l)
        int i = vv + 8 * half;
        float p = (jcol < HW) ? __expf(s[vv] - mrow[i]) * lrow[i] : 0.0f;
        tbt[wv][i][hj * 16 + l16] = (_Float16)p;            // store P[i][j]
      }
    }
    v16h pb = cat8(ld8(&tbt[wv][l16][half * 8]),
                   ld8(&tbt[wv][l16][16 + half * 8]));      // B frag of P^T
#pragma unroll
    for (int t = 0; t < 4; ++t) {
      if (t < ntile) {
        int c0 = cbase + t * 16;
        const _Float16* vr = &vc[c0 + l16][0];              // A frag of V (M=c, K=j)
        v16h va = cat8(ld8(vr + half * 8), ld8(vr + 16 + half * 8));
        acc[t] = WMMA(va, pb, acc[t]);
      }
    }
    __syncthreads();
  }
#pragma unroll
  for (int t = 0; t < 4; ++t) {
    if (t < ntile) {
      int c0 = cbase + t * 16;
#pragma unroll
      for (int vv = 0; vv < 8; ++vv) {
        int c = c0 + vv + 8 * half;
        out[(b * C + c) * HW + i0 + l16] = acc[t][vv];
      }
    }
  }
}

// --------------------------------- residual + instance-norm + leaky + style
__global__ __launch_bounds__(256) void k_norm(
    const float* __restrict__ xc, const float* __restrict__ att,
    const float* __restrict__ gamma_p, const float* __restrict__ styles,
    float* __restrict__ y, int C, int HW, int layer) {
  __shared__ float sred[256], s2red[256];
  const int bc = blockIdx.x;
  const int b  = bc / C;
  const int tid = threadIdx.x;
  const float g = gamma_p[0];
  const float* xp = xc + (size_t)bc * HW;
  const float* ap = att + (size_t)bc * HW;
  float xs[16];
  float s = 0.0f, s2 = 0.0f;
  int cnt = 0;
  for (int t = tid; t < HW; t += 256) {
    float xv = g * ap[t] + xp[t];
    xs[cnt++] = xv;
    s += xv; s2 += xv * xv;
  }
  sred[tid] = s; s2red[tid] = s2;
  __syncthreads();
  for (int st = 128; st > 0; st >>= 1) {
    if (tid < st) { sred[tid] += sred[tid + st]; s2red[tid] += s2red[tid + st]; }
    __syncthreads();
  }
  float mu  = sred[0] / (float)HW;
  float var = s2red[0] / (float)HW - mu * mu;
  float rs  = rsqrtf(var + 1e-5f);
  float sc  = 1.0f + styles[b * 5 + layer];
  float* yp = y + (size_t)bc * HW;
  cnt = 0;
  for (int t = tid; t < HW; t += 256) {
    float xv = (xs[cnt++] - mu) * rs;
    xv = (xv >= 0.0f) ? xv : 0.2f * xv;
    yp[t] = xv * sc;
  }
}

// -------------------------------------- bilinear x2 (jax half-pixel weights)
__global__ void k_upsample(const float* __restrict__ x, float* __restrict__ y,
                           int C, int H, int W) {
  const int OH = 2 * H, OW = 2 * W;
  long total = (long)8 * C * OH * OW;
  long i = (long)blockIdx.x * blockDim.x + threadIdx.x;
  if (i >= total) return;
  int ox = (int)(i % OW); long r = i / OW;
  int oy = (int)(r % OH); r /= OH;                          // r = b*C + c
  const float* xp = x + r * (long)H * W;
  int iy = oy >> 1, ix = ox >> 1;
  int y0, y1, x0, x1; float wy0, wy1, wx0, wx1;
  if (oy & 1) { y0 = iy; y1 = (iy + 1 < H) ? iy + 1 : H - 1; wy0 = 0.75f; wy1 = 0.25f; }
  else        { y0 = (iy > 0) ? iy - 1 : 0; y1 = iy;         wy0 = 0.25f; wy1 = 0.75f; }
  if (ox & 1) { x0 = ix; x1 = (ix + 1 < W) ? ix + 1 : W - 1; wx0 = 0.75f; wx1 = 0.25f; }
  else        { x0 = (ix > 0) ? ix - 1 : 0; x1 = ix;         wx0 = 0.25f; wx1 = 0.75f; }
  y[i] = wy0 * (wx0 * xp[y0 * W + x0] + wx1 * xp[y0 * W + x1]) +
         wy1 * (wx0 * xp[y1 * W + x0] + wx1 * xp[y1 * W + x1]);
}

// ---------------------------------------------------------------- to_rgb
__global__ void k_torgb(const float* __restrict__ x, const float* __restrict__ w,
                        const float* __restrict__ bias, float* __restrict__ out,
                        int C, int HW) {
  int i = blockIdx.x * blockDim.x + threadIdx.x;
  if (i >= 8 * HW) return;
  int b = i / HW, hw = i % HW;
  const float* xp = x + (size_t)(b * C) * HW + hw;
  float acc = bias[0];
  for (int c = 0; c < C; ++c) acc += w[c] * xp[(size_t)c * HW];
  out[i] = acc;
}

// ---------------------------------------------------------------- host
extern "C" void kernel_launch(void* const* d_in, const int* in_sizes, int n_in,
                              void* d_out, int out_size, void* d_ws, size_t ws_size,
                              hipStream_t stream) {
  (void)in_sizes; (void)n_in; (void)out_size; (void)ws_size;
  const float* styles = (const float*)d_in[0];
  const float* cst    = (const float*)d_in[1];
  // per-layer params: 2 + L*9 + {0:conv_w 1:conv_b 2:q_w 3:q_b 4:k_w 5:k_b 6:v_w 7:v_b 8:gamma}
  auto P = [&](int L, int j) { return (const float*)d_in[2 + L * 9 + j]; };
  const float* rgb_w = (const float*)d_in[2 + 45];
  const float* rgb_b = (const float*)d_in[2 + 46];

  static const int IN_C[5]  = {512, 512, 512, 512, 256};
  static const int OUT_C[5] = {512, 512, 512, 256, 128};
  static const int S[5]     = {4, 8, 16, 32, 64};
  const int B = 8;

  // workspace carve (floats): 2 act ping-pong + q + k + v + att  (~100 MiB)
  const size_t ACT = (size_t)8 * 512 * 64 * 64;   // 8,388,608
  const size_t QKB = (size_t)524288;              // max B*(C/8)*HW
  const size_t VB  = (size_t)4194304;             // max B*C*HW
  float* pA   = (float*)d_ws;
  float* pB   = pA + ACT;
  float* bq   = pB + ACT;
  float* bk   = bq + QKB;
  float* bv   = bk + QKB;
  float* batt = bv + VB;

  {
    int npb = 512 * 16, tot = npb * B;
    k_bcast<<<dim3((tot + 255) / 256), 256, 0, stream>>>(cst, pA, npb, tot);
  }
  float* cur = pA;
  float* alt = pB;
  for (int L = 0; L < 5; ++L) {
    const int ic = IN_C[L], oc = OUT_C[L], s = S[L];
    const int HW = s * s, N = B * HW, c8 = oc / 8;
    // conv3x3 (implicit-GEMM WMMA): cur -> alt
    k_gemm<true><<<dim3(oc / 16, N / 64), 128, 0, stream>>>(
        P(L, 0), cur, P(L, 1), alt, oc, ic * 9, HW, s, s, ic);
    // q, k, v 1x1 GEMMs from conv output
    k_gemm<false><<<dim3(c8 / 16, N / 64), 128, 0, stream>>>(
        P(L, 2), alt, P(L, 3), bq, c8, oc, HW, s, s, oc);
    k_gemm<false><<<dim3(c8 / 16, N / 64), 128, 0, stream>>>(
        P(L, 4), alt, P(L, 5), bk, c8, oc, HW, s, s, oc);
    k_gemm<false><<<dim3(oc / 16, N / 64), 128, 0, stream>>>(
        P(L, 6), alt, P(L, 7), bv, oc, oc, HW, s, s, oc);
    // fused flash attention
    k_attn<<<dim3(HW / 16, B), 256, 0, stream>>>(bq, bk, bv, batt, oc, c8, HW);
    // residual + instance norm + leaky + style: -> cur (input is consumed)
    k_norm<<<dim3(B * oc), 256, 0, stream>>>(alt, batt, P(L, 8), styles, cur,
                                             oc, HW, L);
    if (L < 4) {
      long tot = (long)B * oc * 4 * HW;
      k_upsample<<<dim3((int)((tot + 255) / 256)), 256, 0, stream>>>(cur, alt,
                                                                     oc, s, s);
      float* t = cur; cur = alt; alt = t;
    }
  }
  k_torgb<<<dim3((8 * 4096 + 255) / 256), 256, 0, stream>>>(
      cur, rgb_w, rgb_b, (float*)d_out, 128, 4096);
}
